// ChunkwiseNeuralMemory_6012954214665
// MI455X (gfx1250) — matrix operand order repaired and checked by reading, not compiled, not run on hardware
//
#include <hip/hip_runtime.h>
#include <cstdint>
#include <cstddef>

// ---------------------------------------------------------------------------
// Problem constants (fixed by the reference harness)
// ---------------------------------------------------------------------------
constexpr int Bz  = 4;
constexpr int Lz  = 2048;
constexpr int Dz  = 2048;
constexpr int Hz  = 2048;
constexpr int CSz = 64;
constexpr int NCH = Lz / CSz;     // 32 chunks
constexpr int Mtok = Bz * Lz;     // 8192 tokens

typedef __bf16 bf16_t;
typedef __attribute__((ext_vector_type(16))) __bf16 v16bf;
typedef __attribute__((ext_vector_type(8)))  float  v8f;

// ---------------------------------------------------------------------------
// Helpers
// ---------------------------------------------------------------------------
__device__ __forceinline__ bf16_t f2bf(float f) {
  union { float f; uint32_t u; } a; a.f = f;
  uint32_t r = a.u + 0x7fffu + ((a.u >> 16) & 1u);   // RNE
  union { unsigned short s; bf16_t h; } o; o.s = (unsigned short)(r >> 16);
  return o.h;
}

// A fragment: 16x32 bf16 tile from row-major A[M,K]; per ISA layout this is
// two contiguous 16B loads per lane (lane m = L&15, half = L>>4).
__device__ __forceinline__ v16bf load_a_frag(const bf16_t* tile, int lda) {
  int lane = threadIdx.x & 31;
  int m = lane & 15, half = lane >> 4;
  const bf16_t* p = tile + (size_t)m * lda + 8 * half;
  union { v16bf v; uint4 q[2]; } r;
  r.q[0] = *(const uint4*)(p);
  r.q[1] = *(const uint4*)(p + 16);
  return r.v;
}
// B fragment: 32x16 bf16 tile from row-major B[K,N]; strided gather per lane
// (lane n = L&15, half = L>>4; k = 8*half+s, then 16+8*half+(s-8)).
__device__ __forceinline__ v16bf load_b_frag(const bf16_t* tile, int ldb) {
  int lane = threadIdx.x & 31;
  int n = lane & 15, half = lane >> 4;
  v16bf r;
#pragma unroll
  for (int s = 0; s < 16; ++s) {
    int k = (s < 8) ? (8 * half + s) : (16 + 8 * half + (s - 8));
    r[s] = tile[(size_t)k * ldb + n];
  }
  return r;
}
#define WMMA_BF16(a, b, c) \
  __builtin_amdgcn_wmma_f32_16x16x32_bf16(false, (a), false, (b), (short)0, (c), false, false)

__device__ __forceinline__ v8f v8f_zero() {
  v8f z = {0.f, 0.f, 0.f, 0.f, 0.f, 0.f, 0.f, 0.f};
  return z;
}

// ---------------------------------------------------------------------------
// K0: f32 -> bf16 elementwise convert (vectorized x4)
// ---------------------------------------------------------------------------
__global__ void cnm_convert_bf16(const float* __restrict__ in, bf16_t* __restrict__ out, int n4) {
  int i = blockIdx.x * blockDim.x + threadIdx.x;
  if (i >= n4) return;
  float4 v = ((const float4*)in)[i];
  union { bf16_t h[4]; uint2 q; } pk;
  pk.h[0] = f2bf(v.x); pk.h[1] = f2bf(v.y); pk.h[2] = f2bf(v.z); pk.h[3] = f2bf(v.w);
  ((uint2*)out)[i] = pk.q;
}

// ---------------------------------------------------------------------------
// K1: transpose-convert: in f32 [R,C] -> out bf16 [C,R]
// ---------------------------------------------------------------------------
__global__ void cnm_transpose_bf16(const float* __restrict__ in, bf16_t* __restrict__ out,
                                   int R, int C) {
  __shared__ float t[32][33];
  int c0 = blockIdx.x * 32, r0 = blockIdx.y * 32;
  for (int i = threadIdx.y; i < 32; i += 8)
    t[i][threadIdx.x] = in[(size_t)(r0 + i) * C + c0 + threadIdx.x];
  __syncthreads();
  for (int i = threadIdx.y; i < 32; i += 8)
    out[(size_t)(c0 + i) * R + r0 + threadIdx.x] = f2bf(t[threadIdx.x][i]);
}

// ---------------------------------------------------------------------------
// K2: init state: W_init [D,H] f32 -> WT[b][h][d] f32 master + bf16 mirror
// ---------------------------------------------------------------------------
__global__ void cnm_init_state(const float* __restrict__ Wi, float* __restrict__ WTf,
                               bf16_t* __restrict__ WTb) {
  __shared__ float t[32][33];
  int c0 = blockIdx.x * 32, r0 = blockIdx.y * 32;  // in dims: R=Dz rows, C=Hz cols
  for (int i = threadIdx.y; i < 32; i += 8)
    t[i][threadIdx.x] = Wi[(size_t)(r0 + i) * Hz + c0 + threadIdx.x];
  __syncthreads();
  for (int i = threadIdx.y; i < 32; i += 8) {
    float v = t[threadIdx.x][i];
    size_t o = (size_t)(c0 + i) * Dz + r0 + threadIdx.x;  // [H][D]
    bf16_t bv = f2bf(v);
#pragma unroll
    for (int b = 0; b < Bz; ++b) {
      WTf[(size_t)b * Hz * Dz + o] = v;
      WTb[(size_t)b * Hz * Dz + o] = bv;
    }
  }
}

// ---------------------------------------------------------------------------
// K3: per-token dot products  lrz[t]=x_t . fc_lr_w ; wdz[t]=x_t . fc_wd_w
// one wave per token, shuffle reduction
// ---------------------------------------------------------------------------
__global__ void cnm_token_dots(const float* __restrict__ x, const float* __restrict__ fcl,
                               const float* __restrict__ fcw,
                               float* __restrict__ lrz, float* __restrict__ wdz) {
  int wave = threadIdx.x >> 5, lane = threadIdx.x & 31;
  int t = blockIdx.x * 8 + wave;
  const float* row = x + (size_t)t * Dz;
  float s1 = 0.f, s2 = 0.f;
  for (int i = lane; i < Dz; i += 32) {
    float xv = row[i];
    s1 += xv * fcl[i];
    s2 += xv * fcw[i];
  }
#pragma unroll
  for (int off = 16; off > 0; off >>= 1) {
    s1 += __shfl_xor(s1, off, 32);
    s2 += __shfl_xor(s2, off, 32);
  }
  if (lane == 0) { lrz[t] = s1; wdz[t] = s2; }
}

// ---------------------------------------------------------------------------
// K4: per-chunk scalars: lr, cumsum(log_wd)=c, wd_cross=exp(c),
//     e = lr*exp(c[63]-c), wd_last = exp(c[63]).  One block (64 thr) / chunk.
// ---------------------------------------------------------------------------
__global__ void cnm_chunk_scalars(const float* __restrict__ lrz, const float* __restrict__ wdz,
                                  const float* __restrict__ lblr, const float* __restrict__ lbwd,
                                  const float* __restrict__ fclb, const float* __restrict__ fcwb,
                                  float* __restrict__ lr_a, float* __restrict__ wdc_a,
                                  float* __restrict__ e_a, float* __restrict__ c_a,
                                  float* __restrict__ wd_last) {
  __shared__ float s[CSz];
  int b = blockIdx.x / NCH, ch = blockIdx.x % NCH;
  int l = threadIdx.x;
  int g = b * Lz + ch * CSz + l;
  float z1 = lrz[g] + fclb[0];
  float lr = expf(lblr[0]) * (1.f / (1.f + expf(-z1)));
  float z2 = wdz[g] + fcwb[0];
  float logsig = fminf(z2, 0.f) - log1pf(expf(-fabsf(z2)));  // stable log_sigmoid
  s[l] = lbwd[0] + logsig;
  __syncthreads();
  // Hillis-Steele inclusive scan over 64
  for (int off = 1; off < CSz; off <<= 1) {
    float v = s[l];
    float add = (l >= off) ? s[l - off] : 0.f;
    __syncthreads();
    s[l] = v + add;
    __syncthreads();
  }
  float c = s[l];
  float c63 = s[CSz - 1];
  lr_a[g] = lr;
  c_a[g] = c;
  wdc_a[g] = expf(c);
  e_a[g] = lr * expf(c63 - c);
  if (l == CSz - 1) wd_last[b * NCH + ch] = expf(c);
}

// ---------------------------------------------------------------------------
// K5: big GEMM  C[Mtok, 2048] = xbf[Mtok, Dz] @ Bm[Dz, 2048]   (bf16 WMMA)
// 256 thr = 8 waves; 128x128 tile; wave = 32(M) x 64(N) = 2x4 wmma tiles.
// K-loop unrolled x2 with two named fragment buffers (ping-pong): compute on
// one buffer while the other's global_load_b128 clause is in flight -- no
// register-rotation moves.  __launch_bounds__(256,1) avoids scratch spills.
// Outputs selectable: bf16 row-major, bf16 transposed [b][N][2048], f32 rm.
// ---------------------------------------------------------------------------
struct QkvFrags { v16bf a[2]; v16bf b[4]; };

__global__ void __launch_bounds__(256, 1)
cnm_gemm_qkv(const bf16_t* __restrict__ A, const bf16_t* __restrict__ Bm,
             bf16_t* __restrict__ outRm, bf16_t* __restrict__ outTr,
             float* __restrict__ outF) {
  int wave = threadIdx.x >> 5, lane = threadIdx.x & 31;
  int half = lane >> 4, nl = lane & 15;
  int m0 = blockIdx.y * 128 + (wave >> 1) * 32;
  int n0 = blockIdx.x * 128 + (wave & 1) * 64;

  v8f acc[2][4];
#pragma unroll
  for (int i = 0; i < 2; ++i)
#pragma unroll
    for (int j = 0; j < 4; ++j) acc[i][j] = v8f_zero();

  const bf16_t* Arow0 = A + (size_t)m0 * Dz;
  const bf16_t* Arow1 = Arow0 + (size_t)16 * Dz;

  auto loadf = [&](QkvFrags& f, int k0) {
    f.a[0] = load_a_frag(Arow0 + k0, Dz);
    f.a[1] = load_a_frag(Arow1 + k0, Dz);
#pragma unroll
    for (int j = 0; j < 4; ++j)
      f.b[j] = load_b_frag(Bm + (size_t)k0 * Hz + n0 + 16 * j, Hz);
  };
  auto comp = [&](const QkvFrags& f) {
#pragma unroll
    for (int j = 0; j < 4; ++j) {
      acc[0][j] = WMMA_BF16(f.a[0], f.b[j], acc[0][j]);
      acc[1][j] = WMMA_BF16(f.a[1], f.b[j], acc[1][j]);
    }
  };

  QkvFrags f0, f1;
  loadf(f0, 0);
  // 64 K-steps total: steps 1..62 in 31 double iterations, step 63 in the tail
  for (int k0 = 32; k0 + 32 < Dz; k0 += 64) {
    loadf(f1, k0);
    comp(f0);
    loadf(f0, k0 + 32);
    comp(f1);
  }
  loadf(f1, Dz - 32);
  comp(f0);
  comp(f1);

#pragma unroll
  for (int i = 0; i < 2; ++i) {
    int mt = m0 + 16 * i;
#pragma unroll
    for (int j = 0; j < 4; ++j) {
      int nt = n0 + 16 * j;
      if (outRm) {
#pragma unroll
        for (int r = 0; r < 8; ++r)
          outRm[(size_t)(mt + 8 * half + r) * Hz + nt + nl] = f2bf(acc[i][j][r]);
      }
      if (outTr) {
        union { bf16_t h[8]; uint4 q; } pk;
#pragma unroll
        for (int r = 0; r < 8; ++r) pk.h[r] = f2bf(acc[i][j][r]);
        int bb = mt >> 11;                       // batch (tile never crosses)
        int l0 = (mt & (Lz - 1)) + 8 * half;
        *(uint4*)(outTr + (size_t)bb * Hz * Lz + (size_t)(nt + nl) * Lz + l0) = pk.q;
      }
      if (outF) {
#pragma unroll
        for (int r = 0; r < 8; ++r)
          outF[(size_t)(mt + 8 * half + r) * Dz + nt + nl] = acc[i][j][r];
      }
    }
  }
}

// ---------------------------------------------------------------------------
// K6: ycoef[b,l] = lr[l] * sum_{m>=l} (k_l . q_m) * exp(c[m]-c[l])
// One block (8 waves) per (b,chunk): 64x64 kq via WMMA, masked LDS row-reduce.
// ---------------------------------------------------------------------------
struct YcFrags { v16bf a; v16bf b[2]; };

__global__ void __launch_bounds__(256, 1)
cnm_ycoef(const bf16_t* __restrict__ Kbf, const bf16_t* __restrict__ QT,
          const float* __restrict__ c_a, const float* __restrict__ lr_a,
          float* __restrict__ out) {
  __shared__ float cs[CSz];
  __shared__ float rs[CSz];
  int b = blockIdx.x / NCH, ch = blockIdx.x % NCH;
  int tid = threadIdx.x;
  if (tid < CSz) { cs[tid] = c_a[b * Lz + ch * CSz + tid]; rs[tid] = 0.f; }
  __syncthreads();

  int wave = tid >> 5, lane = tid & 31, half = lane >> 4, nl = lane & 15;
  int mt = (wave >> 1) * 16;        // l-tile
  int nt0 = (wave & 1) * 32;        // m-tiles (two of them)
  v8f acc[2] = {v8f_zero(), v8f_zero()};

  const bf16_t* Arow = Kbf + ((size_t)b * Lz + ch * CSz + mt) * Hz;
  const bf16_t* Bbase = QT + (size_t)b * Hz * Lz + ch * CSz + nt0;

  auto loadf = [&](YcFrags& f, int h0) {
    f.a = load_a_frag(Arow + h0, Hz);
    f.b[0] = load_b_frag(Bbase + (size_t)h0 * Lz, Lz);
    f.b[1] = load_b_frag(Bbase + (size_t)h0 * Lz + 16, Lz);
  };
  auto comp = [&](const YcFrags& f) {
    acc[0] = WMMA_BF16(f.a, f.b[0], acc[0]);
    acc[1] = WMMA_BF16(f.a, f.b[1], acc[1]);
  };

  YcFrags f0, f1;
  loadf(f0, 0);
  for (int h0 = 32; h0 + 32 < Hz; h0 += 64) {
    loadf(f1, h0);
    comp(f0);
    loadf(f0, h0 + 32);
    comp(f1);
  }
  loadf(f1, Hz - 32);
  comp(f0);
  comp(f1);

#pragma unroll
  for (int jj = 0; jj < 2; ++jj) {
    int col = nt0 + 16 * jj + nl;   // m index
#pragma unroll
    for (int r = 0; r < 8; ++r) {
      int row = mt + 8 * half + r;  // l index
      if (col >= row) atomicAdd(&rs[row], acc[jj][r] * expf(cs[col] - cs[row]));
    }
  }
  __syncthreads();
  if (tid < CSz) {
    int g = b * Lz + ch * CSz + tid;
    out[g] = lr_a[g] * rs[tid];
  }
}

// ---------------------------------------------------------------------------
// K7 (sequential): WpK[64,D] = K_chunk[64,H] @ WTbf[H,D]; fused epilogue:
//   y = WpK*wd_cross - (WpK - V)*ycoef ;  U = e*(WpK - V)  (bf16)
// 128 thr = 4 waves; tile 64(M) x 64(N); wave = 16 rows x 4 n-tiles.
// Ping-pong pipelined K-loop (latency-critical: ~2 GFLOP per chunk step).
// ---------------------------------------------------------------------------
struct FwdFrags { v16bf a; v16bf b[4]; };

__global__ void __launch_bounds__(128, 1)
cnm_chunk_forward(const bf16_t* __restrict__ Kbf, const bf16_t* __restrict__ WTb,
                  const float* __restrict__ Vf, const float* __restrict__ ycoef,
                  const float* __restrict__ wdc, const float* __restrict__ ea,
                  float* __restrict__ y, bf16_t* __restrict__ U, int chunk) {
  int b = blockIdx.y;
  int wave = threadIdx.x >> 5, lane = threadIdx.x & 31;
  int half = lane >> 4, nl = lane & 15;
  int n0 = blockIdx.x * 64;

  v8f acc[4];
#pragma unroll
  for (int j = 0; j < 4; ++j) acc[j] = v8f_zero();

  const bf16_t* Arow = Kbf + ((size_t)b * Lz + chunk * CSz + wave * 16) * Hz;
  const bf16_t* Bb = WTb + (size_t)b * Hz * Dz + n0;

  auto loadf = [&](FwdFrags& f, int k0) {
    f.a = load_a_frag(Arow + k0, Hz);
#pragma unroll
    for (int j = 0; j < 4; ++j)
      f.b[j] = load_b_frag(Bb + (size_t)k0 * Dz + 16 * j, Dz);
  };
  auto comp = [&](const FwdFrags& f) {
#pragma unroll
    for (int j = 0; j < 4; ++j) acc[j] = WMMA_BF16(f.a, f.b[j], acc[j]);
  };

  FwdFrags f0, f1;
  loadf(f0, 0);
  for (int k0 = 32; k0 + 32 < Hz; k0 += 64) {
    loadf(f1, k0);
    comp(f0);
    loadf(f0, k0 + 32);
    comp(f1);
  }
  loadf(f1, Hz - 32);
  comp(f0);
  comp(f1);

  // per-row scalars (l depends on r only)
  float yc[8], wc[8], er[8];
#pragma unroll
  for (int r = 0; r < 8; ++r) {
    int g = b * Lz + chunk * CSz + wave * 16 + 8 * half + r;
    yc[r] = ycoef[g]; wc[r] = wdc[g]; er[r] = ea[g];
  }
#pragma unroll
  for (int j = 0; j < 4; ++j) {
    int d = n0 + 16 * j + nl;
#pragma unroll
    for (int r = 0; r < 8; ++r) {
      int l = wave * 16 + 8 * half + r;
      size_t g = (size_t)(b * Lz + chunk * CSz + l);
      float wpk = acc[j][r];
      float dv = wpk - Vf[g * Dz + d];
      y[g * Dz + d] = wpk * wc[r] - dv * yc[r];
      U[(size_t)(b * CSz + l) * Dz + d] = f2bf(er[r] * dv);
    }
  }
}

// ---------------------------------------------------------------------------
// K8 (sequential): rank-64 state update
//   WT[h,d] = wd_last * WT[h,d] - sum_l KT[h,l] * U[l,d]
// 256 thr = 8 waves; 128x128 tile; K-loop of 2 (fully unrolled).
// Writes f32 master + bf16 mirror (both L2-resident).
// ---------------------------------------------------------------------------
__global__ void __launch_bounds__(256, 1)
cnm_chunk_update(const bf16_t* __restrict__ KT, const bf16_t* __restrict__ U,
                 const float* __restrict__ wd_last,
                 float* __restrict__ WTf, bf16_t* __restrict__ WTb, int chunk) {
  int b = blockIdx.z;
  int wave = threadIdx.x >> 5, lane = threadIdx.x & 31;
  int half = lane >> 4, nl = lane & 15;
  int h0 = blockIdx.y * 128 + (wave >> 1) * 32;
  int d0 = blockIdx.x * 128 + (wave & 1) * 64;

  v8f acc[2][4];
#pragma unroll
  for (int i = 0; i < 2; ++i)
#pragma unroll
    for (int j = 0; j < 4; ++j) acc[i][j] = v8f_zero();

#pragma unroll
  for (int k0 = 0; k0 < CSz; k0 += 32) {
    v16bf a0 = load_a_frag(KT + (size_t)b * Hz * Lz + (size_t)h0 * Lz + chunk * CSz + k0, Lz);
    v16bf a1 = load_a_frag(KT + (size_t)b * Hz * Lz + (size_t)(h0 + 16) * Lz + chunk * CSz + k0, Lz);
#pragma unroll
    for (int j = 0; j < 4; ++j) {
      v16bf bfr = load_b_frag(U + (size_t)(b * CSz + k0) * Dz + d0 + 16 * j, Dz);
      acc[0][j] = WMMA_BF16(a0, bfr, acc[0][j]);
      acc[1][j] = WMMA_BF16(a1, bfr, acc[1][j]);
    }
  }
  float wd = wd_last[b * NCH + chunk];
#pragma unroll
  for (int i = 0; i < 2; ++i) {
#pragma unroll
    for (int j = 0; j < 4; ++j) {
#pragma unroll
      for (int r = 0; r < 8; ++r) {
        int h = h0 + 16 * i + 8 * half + r;
        int d = d0 + 16 * j + nl;
        size_t idx = (size_t)b * Hz * Dz + (size_t)h * Dz + d;
        float nv = wd * WTf[idx] - acc[i][j][r];
        WTf[idx] = nv;
        WTb[idx] = f2bf(nv);
      }
    }
  }
}

// ---------------------------------------------------------------------------
// Host launcher
// ---------------------------------------------------------------------------
extern "C" void kernel_launch(void* const* d_in, const int* in_sizes, int n_in,
                              void* d_out, int out_size, void* d_ws, size_t ws_size,
                              hipStream_t stream) {
  (void)in_sizes; (void)n_in; (void)out_size; (void)ws_size;
  const float* x     = (const float*)d_in[0];
  const float* lblr  = (const float*)d_in[1];
  const float* lbwd  = (const float*)d_in[2];
  const float* Wq    = (const float*)d_in[3];
  const float* Wk    = (const float*)d_in[4];
  const float* Wv    = (const float*)d_in[5];
  const float* fclw  = (const float*)d_in[6];
  const float* fclb  = (const float*)d_in[7];
  const float* fcww  = (const float*)d_in[8];
  const float* fcwb  = (const float*)d_in[9];
  const float* Winit = (const float*)d_in[10];
  float* y = (float*)d_out;

  // workspace carve-up (256B aligned slabs)
  char* w = (char*)d_ws;
  auto alloc = [&](size_t bytes) -> char* {
    char* p = w; w += (bytes + 255) & ~(size_t)255; return p;
  };
  bf16_t* xbf  = (bf16_t*)alloc((size_t)Mtok * Dz * 2);
  bf16_t* wqT  = (bf16_t*)alloc((size_t)Dz * Hz * 2);
  bf16_t* wkT  = (bf16_t*)alloc((size_t)Dz * Hz * 2);
  bf16_t* wvT  = (bf16_t*)alloc((size_t)Dz * Dz * 2);
  bf16_t* QT   = (bf16_t*)alloc((size_t)Bz * Hz * Lz * 2);
  bf16_t* Kbf  = (bf16_t*)alloc((size_t)Mtok * Hz * 2);
  bf16_t* KT   = (bf16_t*)alloc((size_t)Bz * Hz * Lz * 2);
  float*  Vf   = (float*)alloc((size_t)Mtok * Dz * 4);
  float*  WTf  = (float*)alloc((size_t)Bz * Hz * Dz * 4);
  bf16_t* WTb  = (bf16_t*)alloc((size_t)Bz * Hz * Dz * 2);
  bf16_t* U    = (bf16_t*)alloc((size_t)Bz * CSz * Dz * 2);
  float*  lrz  = (float*)alloc((size_t)Mtok * 4);
  float*  wdz  = (float*)alloc((size_t)Mtok * 4);
  float*  lr_a = (float*)alloc((size_t)Mtok * 4);
  float*  wdc_a= (float*)alloc((size_t)Mtok * 4);
  float*  e_a  = (float*)alloc((size_t)Mtok * 4);
  float*  c_a  = (float*)alloc((size_t)Mtok * 4);
  float*  ycoef= (float*)alloc((size_t)Mtok * 4);
  float*  wdl  = (float*)alloc((size_t)Bz * NCH * 4);

  // Phase 0: conversions / transposes / state init
  int n4 = (Mtok * Dz) / 4;
  cnm_convert_bf16<<<(n4 + 255) / 256, 256, 0, stream>>>(x, xbf, n4);
  dim3 tb(32, 8);
  cnm_transpose_bf16<<<dim3(Dz / 32, Hz / 32), tb, 0, stream>>>(Wq, wqT, Hz, Dz);
  cnm_transpose_bf16<<<dim3(Dz / 32, Hz / 32), tb, 0, stream>>>(Wk, wkT, Hz, Dz);
  cnm_transpose_bf16<<<dim3(Dz / 32, Dz / 32), tb, 0, stream>>>(Wv, wvT, Dz, Dz);
  cnm_init_state<<<dim3(Hz / 32, Dz / 32), tb, 0, stream>>>(Winit, WTf, WTb);

  // Phase 1: scalars
  cnm_token_dots<<<Mtok / 8, 256, 0, stream>>>(x, fclw, fcww, lrz, wdz);
  cnm_chunk_scalars<<<Bz * NCH, CSz, 0, stream>>>(lrz, wdz, lblr, lbwd, fclb, fcwb,
                                                  lr_a, wdc_a, e_a, c_a, wdl);

  // Phase 2: Q/K/V GEMMs (WMMA). Q -> transposed only; K -> rm + transposed; V -> f32.
  dim3 gq(Hz / 128, Mtok / 128);
  cnm_gemm_qkv<<<gq, 256, 0, stream>>>(xbf, wqT, nullptr, QT, nullptr);
  cnm_gemm_qkv<<<gq, 256, 0, stream>>>(xbf, wkT, Kbf, KT, nullptr);
  cnm_gemm_qkv<<<gq, 256, 0, stream>>>(xbf, wvT, nullptr, nullptr, Vf);

  // Phase 3: masked kq row-sums for all chunks in parallel
  cnm_ycoef<<<Bz * NCH, 256, 0, stream>>>(Kbf, QT, c_a, lr_a, ycoef);

  // Phase 4: sequential chunk recurrence (W state L2-resident)
  for (int ch = 0; ch < NCH; ++ch) {
    cnm_chunk_forward<<<dim3(Dz / 64, Bz), 128, 0, stream>>>(
        Kbf, WTb, Vf, ycoef, wdc_a, e_a, y, U, ch);
    cnm_chunk_update<<<dim3(Dz / 128, Hz / 128, Bz), 256, 0, stream>>>(
        KT, U, wdl, WTf, WTb, ch);
  }
}
